// MolGNN_74852690035285
// MI455X (gfx1250) — matrix operand order repaired and checked by reading, not compiled, not run on hardware
//
#include <hip/hip_runtime.h>

#define GNUM 128
#define NPG  512
#define NTOT (GNUM * NPG)     // 65536 nodes
#define EPG  2048
#define ETOT (GNUM * EPG)     // 262144 edges
#define DIN  64
#define HDIM 256
#define LNUM 4
#define DOUTC 10

typedef __attribute__((ext_vector_type(16))) __bf16 v16bf;
typedef __attribute__((ext_vector_type(8)))  __bf16 v8bf;
typedef __attribute__((ext_vector_type(8)))  float  v8f;

// ---------- helpers ----------
__device__ __forceinline__ __bf16 f2bf(float f) {
    unsigned u = __float_as_uint(f);
    unsigned r = (u + 0x7FFFu + ((u >> 16) & 1u)) >> 16;   // RNE
    unsigned short s = (unsigned short)r;
    return __builtin_bit_cast(__bf16, s);
}
__device__ __forceinline__ float bf2f(__bf16 b) {
    unsigned short s = __builtin_bit_cast(unsigned short, b);
    return __uint_as_float(((unsigned)s) << 16);
}

// ---------- degree / norm ----------
__global__ void zero_f(float* p, int n) {
    int i = blockIdx.x * 256 + threadIdx.x;
    if (i < n) p[i] = 0.f;
}

__global__ void edge_deg(const int* __restrict__ row, const int* __restrict__ col,
                         float* __restrict__ deg_sl, float* __restrict__ deg) {
    int e = blockIdx.x * 256 + threadIdx.x;
    if (e < ETOT) {
        atomicAdd(deg_sl + col[e], 1.f);
        atomicAdd(deg + row[e], 1.f);
    }
}

__global__ void finish_dinv(const float* __restrict__ deg_sl, float* __restrict__ dinv) {
    int i = blockIdx.x * 256 + threadIdx.x;
    if (i < NTOT) {
        float d = deg_sl[i] + 1.f;   // self loop
        dinv[i] = rsqrtf(d);
    }
}

// ---------- per-graph stable descending argsort of degree -> unmask layer ----------
__global__ void sort_kernel(const float* __restrict__ deg, int* __restrict__ layer_assign) {
    __shared__ unsigned key[NPG];
    int g = blockIdx.x;
    int t = threadIdx.x;
    unsigned d = (unsigned)deg[g * NPG + t];
    key[t] = (d << 16) | (unsigned)(NPG - 1 - t);   // ties -> smaller index first
    __syncthreads();
    for (int k = 2; k <= NPG; k <<= 1) {
        for (int j = k >> 1; j > 0; j >>= 1) {
            int ixj = t ^ j;
            if (ixj > t) {
                unsigned a = key[t], b = key[ixj];
                bool desc = ((t & k) == 0);
                if (desc ? (a < b) : (a > b)) { key[t] = b; key[ixj] = a; }
            }
            __syncthreads();
        }
    }
    int node = (NPG - 1) - (int)(key[t] & 0xFFFFu);
    layer_assign[g * NPG + node] = t >> 7;   // rank / 128 -> layer in [0,3]
}

// ---------- f32 -> bf16 hi/lo split ----------
__global__ void split_bf16(const float* __restrict__ x, __bf16* __restrict__ hi,
                           __bf16* __restrict__ lo, int n) {
    int i = blockIdx.x * 256 + threadIdx.x;
    if (i < n) {
        float v = x[i];
        __bf16 h = f2bf(v);
        hi[i] = h;
        lo[i] = f2bf(v - bf2f(h));
    }
}

// ---------- pack W (K x 256, row-major) into WMMA B-fragment order ----------
// flat idx = ((ctile*ksteps + ks)*32 + lane)*16 + e
// element = W[ks*32 + (lane/16)*16 + e][ctile*16 + lane%16]
__global__ void pack_b(const float* __restrict__ W, __bf16* __restrict__ Bhi,
                       __bf16* __restrict__ Blo, int K) {
    int idx = blockIdx.x * 256 + threadIdx.x;
    int ksteps = K >> 5;
    int total = 16 * ksteps * 512;
    if (idx >= total) return;
    int e    = idx & 15;
    int lane = (idx >> 4) & 31;
    int blk  = idx >> 9;
    int ks   = blk % ksteps;
    int ct   = blk / ksteps;
    int k = ks * 32 + (lane >> 4) * 16 + e;
    int c = ct * 16 + (lane & 15);
    float v = W[k * HDIM + c];
    __bf16 h = f2bf(v);
    Bhi[idx] = h;
    Blo[idx] = f2bf(v - bf2f(h));
}

// ---------- WMMA GEMM: C[M x 256] = A[M x K] * B[K x 256] (+bias) ----------
// grid = (M/128, 4), block = 256 (8 waves). Wave w -> rows blk*128 + w*16,
// 4 column tiles of 16. bf16x3 split product, f32 accumulate.
__global__ __launch_bounds__(256) void gemm_wmma(
    const __bf16* __restrict__ Ahi, const __bf16* __restrict__ Alo,
    const __bf16* __restrict__ Bhi, const __bf16* __restrict__ Blo,
    const float* __restrict__ bias, float* __restrict__ C, int K) {
    int wid  = threadIdx.x >> 5;
    int lane = threadIdx.x & 31;
    int rowBase = blockIdx.x * 128 + wid * 16;
    int ctile0  = blockIdx.y * 4;
    int ksteps  = K >> 5;
    int r    = lane & 15;
    int half = lane >> 4;
    const __bf16* aHiRow = Ahi + (size_t)(rowBase + r) * K;
    const __bf16* aLoRow = Alo + (size_t)(rowBase + r) * K;

    v8f acc0 = {}, acc1 = {}, acc2 = {}, acc3 = {};

    for (int ks = 0; ks < ksteps; ks++) {
        int base1 = ks * 32 + half * 16;
        v8bf ah0 = *(const v8bf*)(aHiRow + base1);
        v8bf ah1 = *(const v8bf*)(aHiRow + base1 + 8);
        v8bf al0 = *(const v8bf*)(aLoRow + base1);
        v8bf al1 = *(const v8bf*)(aLoRow + base1 + 8);
        v16bf ahi = __builtin_shufflevector(ah0, ah1, 0,1,2,3,4,5,6,7,8,9,10,11,12,13,14,15);
        v16bf alo = __builtin_shufflevector(al0, al1, 0,1,2,3,4,5,6,7,8,9,10,11,12,13,14,15);

#pragma unroll
        for (int t = 0; t < 4; t++) {
            size_t boff = ((size_t)((ctile0 + t) * ksteps + ks) * 32 + lane) * 16;
            v16bf bhi = *(const v16bf*)(Bhi + boff);
            v16bf blo = *(const v16bf*)(Blo + boff);
            v8f acc = (t == 0) ? acc0 : (t == 1) ? acc1 : (t == 2) ? acc2 : acc3;
            acc = __builtin_amdgcn_wmma_f32_16x16x32_bf16(false, ahi, false, bhi,
                                                          (short)0, acc, false, false);
            acc = __builtin_amdgcn_wmma_f32_16x16x32_bf16(false, alo, false, bhi,
                                                          (short)0, acc, false, false);
            acc = __builtin_amdgcn_wmma_f32_16x16x32_bf16(false, ahi, false, blo,
                                                          (short)0, acc, false, false);
            if (t == 0) acc0 = acc; else if (t == 1) acc1 = acc;
            else if (t == 2) acc2 = acc; else acc3 = acc;
        }
    }

#pragma unroll
    for (int t = 0; t < 4; t++) {
        v8f acc = (t == 0) ? acc0 : (t == 1) ? acc1 : (t == 2) ? acc2 : acc3;
        int c = (ctile0 + t) * 16 + (lane & 15);
        float bv = bias ? bias[c] : 0.f;
#pragma unroll
        for (int j = 0; j < 8; j++) {
            int m = rowBase + j + half * 8;
            C[(size_t)m * HDIM + c] = acc[j] + bv;
        }
    }
}

// ---------- aggregation: agg = segsum(hw[row]*norm, col) + self + bias ----------
__global__ void agg_init(const float* __restrict__ hw, const float* __restrict__ dinv,
                         const float* __restrict__ bias, float* __restrict__ agg) {
    int idx = blockIdx.x * 256 + threadIdx.x;   // NTOT*HDIM threads
    int node = idx >> 8;
    int f    = idx & 255;
    float di = dinv[node];
    agg[idx] = hw[idx] * di * di + bias[f];
}

__global__ void agg_edges(const int* __restrict__ row, const int* __restrict__ col,
                          const float* __restrict__ dinv, const float* __restrict__ hw,
                          float* __restrict__ agg) {
    int idx = blockIdx.x * 256 + threadIdx.x;   // ETOT*64 threads
    int e  = idx >> 6;
    int f4 = (idx & 63) << 2;
    int r = row[e], c = col[e];
    float w = dinv[r] * dinv[c];
    const float4 v = *(const float4*)(hw + (size_t)r * HDIM + f4);
    float* dst = agg + (size_t)c * HDIM + f4;
    atomicAdd(dst + 0, v.x * w);
    atomicAdd(dst + 1, v.y * w);
    atomicAdd(dst + 2, v.z * w);
    atomicAdd(dst + 3, v.w * w);
}

// ---------- relu + layernorm + centrality mask (wave per node) ----------
__global__ void post_kernel(const float* __restrict__ agg, float* __restrict__ h,
                            const int* __restrict__ layer_assign,
                            const float* __restrict__ lng, const float* __restrict__ lnb,
                            int l, int do_ln) {
    int node = blockIdx.x * 8 + (threadIdx.x >> 5);
    int lane = threadIdx.x & 31;
    const float* a = agg + (size_t)node * HDIM;
    float v[8];
    float s = 0.f, sq = 0.f;
#pragma unroll
    for (int i = 0; i < 8; i++) {
        float x = a[lane + i * 32];
        if (do_ln) x = fmaxf(x, 0.f);
        v[i] = x; s += x; sq += x * x;
    }
    if (do_ln) {
        for (int off = 16; off > 0; off >>= 1) {
            s  += __shfl_xor(s, off, 32);
            sq += __shfl_xor(sq, off, 32);
        }
        float mu  = s * (1.f / 256.f);
        float var = sq * (1.f / 256.f) - mu * mu;
        float rs  = rsqrtf(var + 1e-5f);
#pragma unroll
        for (int i = 0; i < 8; i++) {
            int f = lane + i * 32;
            v[i] = (v[i] - mu) * rs * lng[f] + lnb[f];
        }
    }
    if (layer_assign[node] == l) {
        float* hp = h + (size_t)node * HDIM;
#pragma unroll
        for (int i = 0; i < 8; i++) hp[lane + i * 32] = v[i];
    }
}

// ---------- mean pool per graph ----------
__global__ void pool_kernel(const float* __restrict__ h, float* __restrict__ pooled) {
    int g = blockIdx.x;
    int f = threadIdx.x;   // 256
    float s = 0.f;
    const float* base = h + (size_t)g * NPG * HDIM + f;
    for (int n = 0; n < NPG; n++) s += base[(size_t)n * HDIM];
    pooled[g * HDIM + f] = s * (1.f / NPG);
}

// ---------- classifier + log_softmax (one wave per graph) ----------
__global__ void cls_kernel(const float* __restrict__ pooled, const float* __restrict__ Wc,
                           const float* __restrict__ bc, float* __restrict__ out) {
    int g = blockIdx.x;
    int lane = threadIdx.x;   // 32
    float accv[DOUTC];
#pragma unroll
    for (int o = 0; o < DOUTC; o++) accv[o] = 0.f;
    const float* p = pooled + g * HDIM;
    for (int f = lane; f < HDIM; f += 32) {
        float pv = p[f];
        const float* wr = Wc + f * DOUTC;
#pragma unroll
        for (int o = 0; o < DOUTC; o++) accv[o] += pv * wr[o];
    }
    for (int off = 16; off > 0; off >>= 1)
#pragma unroll
        for (int o = 0; o < DOUTC; o++) accv[o] += __shfl_xor(accv[o], off, 32);
    if (lane == 0) {
        float logits[DOUTC];
        float m = -1e30f;
#pragma unroll
        for (int o = 0; o < DOUTC; o++) { logits[o] = accv[o] + bc[o]; m = fmaxf(m, logits[o]); }
        float sum = 0.f;
#pragma unroll
        for (int o = 0; o < DOUTC; o++) sum += expf(logits[o] - m);
        float lse = logf(sum);
#pragma unroll
        for (int o = 0; o < DOUTC; o++) out[g * DOUTC + o] = logits[o] - m - lse;
    }
}

__global__ void ycopy_kernel(const int* __restrict__ y, int* __restrict__ outi) {
    int i = threadIdx.x;
    if (i < GNUM) outi[i] = y[i];
}

// ---------- host orchestration ----------
extern "C" void kernel_launch(void* const* d_in, const int* in_sizes, int n_in,
                              void* d_out, int out_size, void* d_ws, size_t ws_size,
                              hipStream_t stream) {
    (void)in_sizes; (void)n_in; (void)out_size; (void)ws_size;
    const float* x    = (const float*)d_in[0];
    const int*   ei   = (const int*)d_in[1];
    const int*   y    = (const int*)d_in[4];
    const float* W0   = (const float*)d_in[5];
    const float* b0   = (const float*)d_in[6];
    const float* Wl   = (const float*)d_in[7];
    const float* bl   = (const float*)d_in[8];
    const float* ln_g = (const float*)d_in[9];
    const float* ln_b = (const float*)d_in[10];
    const float* Wc   = (const float*)d_in[11];
    const float* bc   = (const float*)d_in[12];
    const int* row = ei;
    const int* col = ei + ETOT;

    char* p = (char*)d_ws;
    auto carve = [&](size_t bytes) -> void* {
        void* r = (void*)p;
        p += (bytes + 255) & ~(size_t)255;
        return r;
    };
    float*  h       = (float*)carve((size_t)NTOT * HDIM * 4);
    float*  hw      = (float*)carve((size_t)NTOT * HDIM * 4);
    float*  agg     = (float*)carve((size_t)NTOT * HDIM * 4);
    __bf16* h_hi    = (__bf16*)carve((size_t)NTOT * HDIM * 2);
    __bf16* h_lo    = (__bf16*)carve((size_t)NTOT * HDIM * 2);
    __bf16* Bp_hi   = (__bf16*)carve((size_t)HDIM * HDIM * 2);
    __bf16* Bp_lo   = (__bf16*)carve((size_t)HDIM * HDIM * 2);
    float*  deg_sl  = (float*)carve((size_t)NTOT * 4);
    float*  deg     = (float*)carve((size_t)NTOT * 4);
    float*  dinv    = (float*)carve((size_t)NTOT * 4);
    int*    lassign = (int*)carve((size_t)NTOT * 4);
    float*  pooled  = (float*)carve((size_t)GNUM * HDIM * 4);

    // degrees + norm
    zero_f<<<(2 * NTOT + 255) / 256, 256, 0, stream>>>(deg_sl, 2 * NTOT); // deg_sl & deg contiguous
    edge_deg<<<(ETOT + 255) / 256, 256, 0, stream>>>(row, col, deg_sl, deg);
    finish_dinv<<<(NTOT + 255) / 256, 256, 0, stream>>>(deg_sl, dinv);
    sort_kernel<<<GNUM, NPG, 0, stream>>>(deg, lassign);

    // h = x @ W0 + b0
    split_bf16<<<(NTOT * DIN + 255) / 256, 256, 0, stream>>>(x, h_hi, h_lo, NTOT * DIN);
    pack_b<<<(16 * (DIN / 32) * 512 + 255) / 256, 256, 0, stream>>>(W0, Bp_hi, Bp_lo, DIN);
    gemm_wmma<<<dim3(NTOT / 128, 4), 256, 0, stream>>>(h_hi, h_lo, Bp_hi, Bp_lo, b0, h, DIN);

    for (int l = 0; l < LNUM; l++) {
        split_bf16<<<(NTOT * HDIM + 255) / 256, 256, 0, stream>>>(h, h_hi, h_lo, NTOT * HDIM);
        pack_b<<<(16 * (HDIM / 32) * 512 + 255) / 256, 256, 0, stream>>>(
            Wl + (size_t)l * HDIM * HDIM, Bp_hi, Bp_lo, HDIM);
        gemm_wmma<<<dim3(NTOT / 128, 4), 256, 0, stream>>>(h_hi, h_lo, Bp_hi, Bp_lo,
                                                           (const float*)nullptr, hw, HDIM);
        agg_init<<<((size_t)NTOT * HDIM + 255) / 256, 256, 0, stream>>>(hw, dinv, bl + l * HDIM, agg);
        agg_edges<<<((size_t)ETOT * 64 + 255) / 256, 256, 0, stream>>>(row, col, dinv, hw, agg);
        int do_ln = (l < LNUM - 1) ? 1 : 0;
        const float* lg = ln_g + (size_t)(do_ln ? l : 0) * HDIM;
        const float* lb = ln_b + (size_t)(do_ln ? l : 0) * HDIM;
        post_kernel<<<NTOT / 8, 256, 0, stream>>>(agg, h, lassign, lg, lb, l, do_ln);
    }

    pool_kernel<<<GNUM, HDIM, 0, stream>>>(h, pooled);
    cls_kernel<<<GNUM, 32, 0, stream>>>(pooled, Wc, bc, (float*)d_out);
    ycopy_kernel<<<1, 128, 0, stream>>>(y, (int*)((float*)d_out + GNUM * DOUTC));
}
